// MirrorSystem_24902220382535
// MI455X (gfx1250) — matrix operand order repaired and checked by reading, not compiled, not run on hardware
//
#include <hip/hip_runtime.h>
#include <math.h>

// ---------------------------------------------------------------------------
// Problem constants (B=8, S=4096, D=512, N=1024)
// ---------------------------------------------------------------------------
#define M_TOT   32768       // B*S rows
#define K_TOT   1024        // 2*D
#define N_TOT   1024        // codebook entries
#define D_HALF  512
#define INV_K   (1.0f / 1024.0f)
#define LAMBDA_ENERGY 0.1f
#define EPS_CFG 1e-6f
#define OUT_IMAG_OFF 16777216   // B*S*D
#define OUT_ENERGY_OFF 33554432 // 2*B*S*D
#define NELEM_TOTAL 33554432.0f // M_TOT*N_TOT
#define LDSTR 40                // padded LDS tile row stride (bf16 elems, 80B)

typedef __attribute__((ext_vector_type(8)))  __bf16 v8bf;
typedef __attribute__((ext_vector_type(16))) __bf16 v16bf;
typedef __attribute__((ext_vector_type(8)))  float  v8f;

// ---------------------------------------------------------------------------
// WMMA fragment gathers from LDS tiles stored as [row][LDSTR] bf16
// (K-contiguous rows) per CDNA5 ISA 7.12.2 layouts (wave32).
// A 16x32 bf16: lanes 0-15 row M=lane with K {0..7,16..23}; lanes 16-31 same
// rows with K {8..15,24..31}.  B 32x16 bf16: lane = column, same K order as A
// (lanes 0-15: K {0..7,16..23}, lanes 16-31: K {8..15,24..31}).
// ---------------------------------------------------------------------------
union FragU { v16bf v; v8bf h[2]; };

__device__ inline v16bf load_a_frag(const __bf16* tile, int rowBase, int lane) {
    const int m  = rowBase + (lane & 15);
    const int kb = (lane < 16) ? 0 : 8;
    const __bf16* p = tile + m * LDSTR + kb;
    FragU f;
    f.h[0] = *(const v8bf*)(p);        // K = kb .. kb+7
    f.h[1] = *(const v8bf*)(p + 16);   // K = kb+16 .. kb+23
    return f.v;
}

__device__ inline v16bf load_b_frag(const __bf16* tile, int colBase, int lane) {
    const int n  = colBase + (lane & 15);
    const int kb = (lane < 16) ? 0 : 8;
    const __bf16* p = tile + n * LDSTR + kb;
    FragU f;
    f.h[0] = *(const v8bf*)(p);        // K = kb .. kb+7
    f.h[1] = *(const v8bf*)(p + 16);   // K = kb+16 .. kb+23
    return f.v;
}

__device__ inline v8bf pack8(float4 a, float4 b) {
    v8bf r = { (__bf16)a.x, (__bf16)a.y, (__bf16)a.z, (__bf16)a.w,
               (__bf16)b.x, (__bf16)b.y, (__bf16)b.z, (__bf16)b.w };
    return r;
}

// ---------------------------------------------------------------------------
// Kernel 1: per-row ||z||^2 (zz) + global sums of |z| and |z|^2 for variance
// ---------------------------------------------------------------------------
__global__ __launch_bounds__(256) void mag_stats_kernel(
    const float* __restrict__ zr, const float* __restrict__ zi,
    float* __restrict__ zz, float* __restrict__ scalars)
{
    __shared__ float s1[256];
    __shared__ float s2[256];
    const int m = blockIdx.x;
    const int tid = threadIdx.x;
    float a = 0.0f, b = 0.0f;
    for (int d = tid; d < D_HALF; d += 256) {
        const float r  = zr[m * D_HALF + d];
        const float im = zi[m * D_HALF + d];
        const float sq = r * r + im * im;
        a += sqrtf(sq);
        b += sq;
    }
    s1[tid] = a; s2[tid] = b;
    __syncthreads();
    for (int off = 128; off > 0; off >>= 1) {
        if (tid < off) { s1[tid] += s1[tid + off]; s2[tid] += s2[tid + off]; }
        __syncthreads();
    }
    if (tid == 0) {
        zz[m] = s2[0];                       // row norm^2 (reused by GEMM1)
        atomicAdd(&scalars[0], s1[0]);       // sum |z|
        atomicAdd(&scalars[1], s2[0]);       // sum |z|^2
    }
}

// ---------------------------------------------------------------------------
// Kernel 2: per-codeword ||c_n||^2
// ---------------------------------------------------------------------------
__global__ __launch_bounds__(256) void cc_kernel(
    const float* __restrict__ codebook, float* __restrict__ cc)
{
    __shared__ float s[256];
    const int n = blockIdx.x;
    const int tid = threadIdx.x;
    float a = 0.0f;
    for (int k = tid; k < K_TOT; k += 256) {
        const float c = codebook[n * K_TOT + k];
        a += c * c;
    }
    s[tid] = a;
    __syncthreads();
    for (int off = 128; off > 0; off >>= 1) {
        if (tid < off) s[tid] += s[tid + off];
        __syncthreads();
    }
    if (tid == 0) cc[n] = s[0];
}

// ---------------------------------------------------------------------------
// Kernel 3: scale = lambda * softplus(var_unbiased(|z|) / (1+eps))
// ---------------------------------------------------------------------------
__global__ void finalize_scale_kernel(float* __restrict__ scalars)
{
    if (threadIdx.x == 0 && blockIdx.x == 0) {
        const float M = 16777216.0f;     // B*S*D elements of |z|
        const float sum = scalars[0];
        const float sumsq = scalars[1];
        const float mean = sum / M;
        const float var = (sumsq - M * mean * mean) / (M - 1.0f);
        const float v = var / (1.0f + EPS_CFG);
        const float sp = log1pf(expf(v));          // softplus (scalar, keep precise)
        scalars[2] = LAMBDA_ENERGY * sp;
    }
}

// ---------------------------------------------------------------------------
// Kernel 4: fused dual GEMM (z·c^T and prev·adj share the K loop) -> d_total
// WG tile: 64 rows x 128 cols, 8 waves (2x4), wave = 32x32 (2x2 WMMA frags).
// LDS staging fully vectorized: float4 global loads -> v8bf pack ->
// single ds_store_b128 per chunk; tile rows padded to 40 elems (20 banks)
// so the 16-lane fragment gathers are bank-conflict free.
// ---------------------------------------------------------------------------
__global__ __launch_bounds__(256) void dtotal_gemm_kernel(
    const float* __restrict__ z_real, const float* __restrict__ z_imag,
    const float* __restrict__ prev,   const float* __restrict__ codebook,
    const float* __restrict__ adj,    const float* __restrict__ zz,
    const float* __restrict__ cc,     const float* __restrict__ scalars,
    float* __restrict__ dtot)
{
    __shared__ alignas(16) __bf16 zA[64 * LDSTR];   // z_flat tile  [r][k]
    __shared__ alignas(16) __bf16 pA[64 * LDSTR];   // prev tile    [r][k]
    __shared__ alignas(16) __bf16 cB[128 * LDSTR];  // codebook B   [n][k]
    __shared__ alignas(16) __bf16 aB[128 * LDSTR];  // adj^T B      [n][k]

    const int tid  = threadIdx.x;
    const int lane = tid & 31;
    const int wid  = tid >> 5;
    const int wrow = wid >> 2;        // 0..1
    const int wcol = wid & 3;         // 0..3
    const int row0 = blockIdx.y * 64;
    const int col0 = blockIdx.x * 128;

    v8f acc1[2][2] = {};
    v8f acc2[2][2] = {};

    for (int k0 = 0; k0 < K_TOT; k0 += 32) {
        __syncthreads();
        // ---- stage A tiles: each thread handles one 8-elem k-chunk ----
        {
            const int r  = tid >> 2;            // 0..63
            const int kc = (tid & 3) * 8;       // 0,8,16,24
            const int m  = row0 + r;
            const int gk = k0 + kc;             // chunk never crosses 512
            const float* zsrc = (gk < D_HALF)
                ? (z_real + m * D_HALF + gk)
                : (z_imag + m * D_HALF + (gk - D_HALF));
            const float4 z0 = *(const float4*)(zsrc);
            const float4 z1 = *(const float4*)(zsrc + 4);
            *(v8bf*)(&zA[r * LDSTR + kc]) = pack8(z0, z1);

            const float* psrc = prev + m * N_TOT + gk;
            const float4 p0 = *(const float4*)(psrc);
            const float4 p1 = *(const float4*)(psrc + 4);
            *(v8bf*)(&pA[r * LDSTR + kc]) = pack8(p0, p1);
        }
        // ---- stage B tiles: 512 chunks, 2 per thread ----
        #pragma unroll
        for (int i = 0; i < 2; ++i) {
            const int ch = i * 256 + tid;
            const int n  = ch >> 2;             // 0..127
            const int kc = (ch & 3) * 8;
            // codebook[n][k] is K-contiguous
            const float* csrc = codebook + (col0 + n) * K_TOT + k0 + kc;
            const float4 c0 = *(const float4*)(csrc);
            const float4 c1 = *(const float4*)(csrc + 4);
            *(v8bf*)(&cB[n * LDSTR + kc]) = pack8(c0, c1);
            // adjacency[k][n]: k-strided gather (adj is L2-resident, 4MB)
            const float* asrc = adj + (k0 + kc) * N_TOT + (col0 + n);
            v8bf ab = { (__bf16)asrc[0 * N_TOT], (__bf16)asrc[1 * N_TOT],
                        (__bf16)asrc[2 * N_TOT], (__bf16)asrc[3 * N_TOT],
                        (__bf16)asrc[4 * N_TOT], (__bf16)asrc[5 * N_TOT],
                        (__bf16)asrc[6 * N_TOT], (__bf16)asrc[7 * N_TOT] };
            *(v8bf*)(&aB[n * LDSTR + kc]) = ab;
        }
        __syncthreads();

        const v16bf a0 = load_a_frag(zA, wrow * 32,      lane);
        const v16bf a1 = load_a_frag(zA, wrow * 32 + 16, lane);
        const v16bf p0 = load_a_frag(pA, wrow * 32,      lane);
        const v16bf p1 = load_a_frag(pA, wrow * 32 + 16, lane);
        const v16bf b0 = load_b_frag(cB, wcol * 32,      lane);
        const v16bf b1 = load_b_frag(cB, wcol * 32 + 16, lane);
        const v16bf g0 = load_b_frag(aB, wcol * 32,      lane);
        const v16bf g1 = load_b_frag(aB, wcol * 32 + 16, lane);

        acc1[0][0] = __builtin_amdgcn_wmma_f32_16x16x32_bf16(false, a0, false, b0, (short)0, acc1[0][0], false, false);
        acc1[0][1] = __builtin_amdgcn_wmma_f32_16x16x32_bf16(false, a0, false, b1, (short)0, acc1[0][1], false, false);
        acc1[1][0] = __builtin_amdgcn_wmma_f32_16x16x32_bf16(false, a1, false, b0, (short)0, acc1[1][0], false, false);
        acc1[1][1] = __builtin_amdgcn_wmma_f32_16x16x32_bf16(false, a1, false, b1, (short)0, acc1[1][1], false, false);
        acc2[0][0] = __builtin_amdgcn_wmma_f32_16x16x32_bf16(false, p0, false, g0, (short)0, acc2[0][0], false, false);
        acc2[0][1] = __builtin_amdgcn_wmma_f32_16x16x32_bf16(false, p0, false, g1, (short)0, acc2[0][1], false, false);
        acc2[1][0] = __builtin_amdgcn_wmma_f32_16x16x32_bf16(false, p1, false, g0, (short)0, acc2[1][0], false, false);
        acc2[1][1] = __builtin_amdgcn_wmma_f32_16x16x32_bf16(false, p1, false, g1, (short)0, acc2[1][1], false, false);
    }

    // ---- epilogue: d_total = d_content - scale*sigmoid(graph_bias) ----
    const float scale = scalars[2];
    #pragma unroll
    for (int ti = 0; ti < 2; ++ti) {
        #pragma unroll
        for (int tj = 0; tj < 2; ++tj) {
            const int mb = row0 + wrow * 32 + ti * 16 + ((lane >= 16) ? 8 : 0);
            const int n  = col0 + wcol * 32 + tj * 16 + (lane & 15);
            const float ccn = cc[n];
            #pragma unroll
            for (int r = 0; r < 8; ++r) {
                const int m = mb + r;
                const float dc = (zz[m] + ccn - 2.0f * acc1[ti][tj][r]) * INV_K;
                const float g  = acc2[ti][tj][r];
                // v_exp_f32 + v_rcp_f32 gating sigmoid
                const float sg = __builtin_amdgcn_rcpf(1.0f + __expf(-g));
                dtot[m * N_TOT + n] = dc - scale * sg;
            }
        }
    }
}

// ---------------------------------------------------------------------------
// Kernel 5: row softmax of -d_total -> probs (bf16); accumulate sum(d_total)
// ---------------------------------------------------------------------------
__global__ __launch_bounds__(256) void softmax_kernel(
    const float* __restrict__ dtot, __bf16* __restrict__ probs,
    float* __restrict__ scalars)
{
    __shared__ float red[256];
    const int m = blockIdx.x;
    const int tid = threadIdx.x;

    float x[4];
    float dsum = 0.0f;
    #pragma unroll
    for (int j = 0; j < 4; ++j) {
        const float d = dtot[m * N_TOT + j * 256 + tid];
        x[j] = -d;
        dsum += d;
    }
    // energy partial
    red[tid] = dsum;
    __syncthreads();
    for (int off = 128; off > 0; off >>= 1) {
        if (tid < off) red[tid] += red[tid + off];
        __syncthreads();
    }
    if (tid == 0) atomicAdd(&scalars[3], red[0]);
    __syncthreads();

    // max
    float mx = fmaxf(fmaxf(x[0], x[1]), fmaxf(x[2], x[3]));
    red[tid] = mx;
    __syncthreads();
    for (int off = 128; off > 0; off >>= 1) {
        if (tid < off) red[tid] = fmaxf(red[tid], red[tid + off]);
        __syncthreads();
    }
    mx = red[0];
    __syncthreads();

    // exp + sum (max-subtracted -> bounded args, hw exp is plenty accurate)
    float e[4];
    float s = 0.0f;
    #pragma unroll
    for (int j = 0; j < 4; ++j) { e[j] = __expf(x[j] - mx); s += e[j]; }
    red[tid] = s;
    __syncthreads();
    for (int off = 128; off > 0; off >>= 1) {
        if (tid < off) red[tid] += red[tid + off];
        __syncthreads();
    }
    const float inv = __builtin_amdgcn_rcpf(red[0]);   // v_rcp_f32
    #pragma unroll
    for (int j = 0; j < 4; ++j)
        probs[m * N_TOT + j * 256 + tid] = (__bf16)(e[j] * inv);
}

// ---------------------------------------------------------------------------
// Kernel 6: z_q = probs x codebook (WMMA), split into real/imag; energy.
// ---------------------------------------------------------------------------
__global__ __launch_bounds__(256) void zq_gemm_kernel(
    const __bf16* __restrict__ probs, const float* __restrict__ codebook,
    const float* __restrict__ scalars, float* __restrict__ out)
{
    __shared__ alignas(16) __bf16 pA[64 * LDSTR];    // probs tile [r][k=n]
    __shared__ alignas(16) __bf16 cB[128 * LDSTR];   // codebook^T tile [d][k=n]

    const int tid  = threadIdx.x;
    const int lane = tid & 31;
    const int wid  = tid >> 5;
    const int wrow = wid >> 2;
    const int wcol = wid & 3;
    const int row0 = blockIdx.y * 64;
    const int col0 = blockIdx.x * 128;   // output feature block (0..1023)

    v8f acc[2][2] = {};

    for (int k0 = 0; k0 < N_TOT; k0 += 32) {
        __syncthreads();
        // probs already bf16: direct 16B copy, one chunk per thread
        {
            const int r  = tid >> 2;
            const int kc = (tid & 3) * 8;
            *(v8bf*)(&pA[r * LDSTR + kc]) =
                *(const v8bf*)(probs + (row0 + r) * N_TOT + k0 + kc);
        }
        // codebook[n][d] -> [d][n-chunk] transpose (codebook is L2-resident)
        #pragma unroll
        for (int i = 0; i < 2; ++i) {
            const int ch = i * 256 + tid;
            const int c  = ch >> 2;             // output feature in tile
            const int kc = (ch & 3) * 8;        // n-chunk
            const float* src = codebook + (k0 + kc) * K_TOT + (col0 + c);
            v8bf b = { (__bf16)src[0 * K_TOT], (__bf16)src[1 * K_TOT],
                       (__bf16)src[2 * K_TOT], (__bf16)src[3 * K_TOT],
                       (__bf16)src[4 * K_TOT], (__bf16)src[5 * K_TOT],
                       (__bf16)src[6 * K_TOT], (__bf16)src[7 * K_TOT] };
            *(v8bf*)(&cB[c * LDSTR + kc]) = b;
        }
        __syncthreads();

        const v16bf a0 = load_a_frag(pA, wrow * 32,      lane);
        const v16bf a1 = load_a_frag(pA, wrow * 32 + 16, lane);
        const v16bf b0 = load_b_frag(cB, wcol * 32,      lane);
        const v16bf b1 = load_b_frag(cB, wcol * 32 + 16, lane);

        acc[0][0] = __builtin_amdgcn_wmma_f32_16x16x32_bf16(false, a0, false, b0, (short)0, acc[0][0], false, false);
        acc[0][1] = __builtin_amdgcn_wmma_f32_16x16x32_bf16(false, a0, false, b1, (short)0, acc[0][1], false, false);
        acc[1][0] = __builtin_amdgcn_wmma_f32_16x16x32_bf16(false, a1, false, b0, (short)0, acc[1][0], false, false);
        acc[1][1] = __builtin_amdgcn_wmma_f32_16x16x32_bf16(false, a1, false, b1, (short)0, acc[1][1], false, false);
    }

    #pragma unroll
    for (int ti = 0; ti < 2; ++ti) {
        #pragma unroll
        for (int tj = 0; tj < 2; ++tj) {
            const int mb = row0 + wrow * 32 + ti * 16 + ((lane >= 16) ? 8 : 0);
            const int d  = col0 + wcol * 32 + tj * 16 + (lane & 15);
            #pragma unroll
            for (int r = 0; r < 8; ++r) {
                const int m = mb + r;
                const float v = acc[ti][tj][r];
                if (d < D_HALF)
                    out[m * D_HALF + d] = v;
                else
                    out[OUT_IMAG_OFF + m * D_HALF + (d - D_HALF)] = v;
            }
        }
    }

    if (blockIdx.x == 0 && blockIdx.y == 0 && tid == 0)
        out[OUT_ENERGY_OFF] = scalars[3] * (1.0f / NELEM_TOTAL);
}

// ---------------------------------------------------------------------------
// Host-side launch.  Workspace layout (bytes):
//   [0,       4096)      scalars: {sum|z|, sum|z|^2, scale, sum d_total}
//   [4096,    135168)    zz[M_TOT]   f32
//   [135168,  139264)    cc[N_TOT]   f32
//   [139264,  67248128)  probs       bf16  (M_TOT*N_TOT)
//   [67248128,201465856) d_total     f32   (M_TOT*N_TOT)
// ---------------------------------------------------------------------------
extern "C" void kernel_launch(void* const* d_in, const int* in_sizes, int n_in,
                              void* d_out, int out_size, void* d_ws, size_t ws_size,
                              hipStream_t stream) {
    const float* z_real   = (const float*)d_in[0];
    const float* z_imag   = (const float*)d_in[1];
    const float* prev     = (const float*)d_in[2];
    const float* codebook = (const float*)d_in[3];
    const float* adj      = (const float*)d_in[4];
    float* out = (float*)d_out;

    char* ws = (char*)d_ws;
    float*  scalars = (float*)(ws);
    float*  zz      = (float*)(ws + 4096);
    float*  cc      = (float*)(ws + 135168);
    __bf16* probs   = (__bf16*)(ws + 139264);
    float*  dtot    = (float*)(ws + 67248128);

    (void)in_sizes; (void)n_in; (void)out_size; (void)ws_size;

    hipMemsetAsync(d_ws, 0, 4096, stream);

    mag_stats_kernel<<<M_TOT, 256, 0, stream>>>(z_real, z_imag, zz, scalars);
    cc_kernel<<<N_TOT, 256, 0, stream>>>(codebook, cc);
    finalize_scale_kernel<<<1, 32, 0, stream>>>(scalars);

    dtotal_gemm_kernel<<<dim3(N_TOT / 128, M_TOT / 64), 256, 0, stream>>>(
        z_real, z_imag, prev, codebook, adj, zz, cc, scalars, dtot);

    softmax_kernel<<<M_TOT, 256, 0, stream>>>(dtot, probs, scalars);

    zq_gemm_kernel<<<dim3(K_TOT / 128, M_TOT / 64), 256, 0, stream>>>(
        probs, codebook, scalars, out);
}
